// Transformer_33380485824619
// MI455X (gfx1250) — matrix operand order repaired and checked by reading, not compiled, not run on hardware
//
#include <hip/hip_runtime.h>
#include <hip/hip_bf16.h>

typedef __attribute__((ext_vector_type(16))) __bf16 v16bf;
typedef __attribute__((ext_vector_type(8)))  __bf16 v8bf;
typedef __attribute__((ext_vector_type(8)))  float  v8f;
typedef __attribute__((ext_vector_type(4)))  int    v4i;

#ifndef __has_builtin
#define __has_builtin(x) 0
#endif
#if __has_builtin(__builtin_amdgcn_global_load_async_to_lds_b128)
#define ASYNC_LDS 1
#define AS1 __attribute__((address_space(1)))
#define AS3 __attribute__((address_space(3)))
#else
#define ASYNC_LDS 0
#endif

__device__ __forceinline__ void wait_async_lds() {
#if ASYNC_LDS
#if __has_builtin(__builtin_amdgcn_s_wait_asynccnt)
  __builtin_amdgcn_s_wait_asynccnt(0);
#else
  asm volatile("s_wait_asynccnt 0" ::: "memory");
#endif
#endif
}

// Problem constants
constexpr int Bb  = 2;
constexpr int Ss  = 2048;
constexpr int Dd  = 512;
constexpr int Hh  = 8;
constexpr int DHd = 64;
constexpr int DCc = 2048;          // DIM_COND
constexpr int Mm  = Bb * Ss;       // 4096 tokens
constexpr int FF   = 1365;         // FF_INNER
constexpr int FF2  = 2730;         // 2*FF_INNER
constexpr int FF2P = 2752;         // padded to 64
constexpr int FFKP = 1376;         // FF_INNER padded to 32

// ---------------------------------------------------------------------------
// Time-conditioning embedding: cond[b][j] = silu(femb @ time_w + time_b)
// femb = [t, sin(2*pi*t*fw), cos(2*pi*t*fw)]  (513 elems)
// ---------------------------------------------------------------------------
__global__ __launch_bounds__(128)
void k_cond(const float* __restrict__ times, const float* __restrict__ fw,
            const float* __restrict__ tw, const float* __restrict__ tb,
            float* __restrict__ cond) {
  int id = blockIdx.x * 128 + threadIdx.x;       // < Bb*DCc = 4096
  int b = id >> 11, j = id & (DCc - 1);
  float t = times[b];
  float acc = t * tw[j];
  for (int i = 0; i < 256; ++i) {
    float f = t * fw[i] * 6.28318530717958647692f;
    acc += __sinf(f) * tw[(size_t)(1 + i) * DCc + j];
    acc += __cosf(f) * tw[(size_t)(257 + i) * DCc + j];
  }
  acc += tb[j];
  cond[id] = acc / (1.f + __expf(-acc));         // silu
}

// ---------------------------------------------------------------------------
// FiLM / zero-gate projections (tiny: B=2 rows)
// ---------------------------------------------------------------------------
__global__ __launch_bounds__(128)
void k_film(const float* __restrict__ cond,
            const float* __restrict__ film_w, const float* __restrict__ film_b,
            const float* __restrict__ zero_w, const float* __restrict__ zero_b,
            float* __restrict__ gamma, float* __restrict__ beta,
            float* __restrict__ gate) {
  int id = blockIdx.x * 128 + threadIdx.x;       // < Bb*1536 = 3072
  int b = id / 1536, j = id % 1536;
  const float* cp = cond + (size_t)b * DCc;
  if (j < 1024) {
    float acc = film_b[j];
    for (int k = 0; k < DCc; ++k) acc += cp[k] * film_w[(size_t)k * 1024 + j];
    if (j < Dd) gamma[b * Dd + j] = acc;
    else        beta[b * Dd + (j - Dd)] = acc;
  } else {
    int jj = j - 1024;
    float acc = zero_b[jj];
    for (int k = 0; k < DCc; ++k) acc += cp[k] * zero_w[(size_t)k * Dd + jj];
    gate[b * Dd + jj] = 1.f / (1.f + __expf(-acc));
  }
}

// ---------------------------------------------------------------------------
// Fused: layernorm -> FiLM-or-ln_g modulation -> rmsnorm(norm_g) -> bf16
// ---------------------------------------------------------------------------
__global__ __launch_bounds__(128)
void k_adaln(const float* __restrict__ x, const unsigned char* __restrict__ mod,
             const float* __restrict__ gamma, const float* __restrict__ beta,
             const float* __restrict__ ln_g, const float* __restrict__ norm_g,
             __bf16* __restrict__ out, int ldo) {
  __shared__ float sh[128];
  const int tok = blockIdx.x, tid = threadIdx.x;
  const int bidx = tok >> 11;                    // token / S
  const float* xp = x + (size_t)tok * Dd;
  float v[4];
  float s = 0.f;
#pragma unroll
  for (int k = 0; k < 4; ++k) { v[k] = xp[tid + 128 * k]; s += v[k]; }
  sh[tid] = s; __syncthreads();
  for (int st = 64; st > 0; st >>= 1) { if (tid < st) sh[tid] += sh[tid + st]; __syncthreads(); }
  float mu = sh[0] * (1.f / Dd); __syncthreads();
  float s2 = 0.f;
#pragma unroll
  for (int k = 0; k < 4; ++k) { float d = v[k] - mu; s2 += d * d; }
  sh[tid] = s2; __syncthreads();
  for (int st = 64; st > 0; st >>= 1) { if (tid < st) sh[tid] += sh[tid + st]; __syncthreads(); }
  float var = sh[0] * (1.f / Dd); __syncthreads();
  float rstd = rsqrtf(var + 1e-5f);
  bool m = mod[tok] != 0;
  float xin[4]; float q = 0.f;
#pragma unroll
  for (int k = 0; k < 4; ++k) {
    int j = tid + 128 * k;
    float hn = (v[k] - mu) * rstd;
    float t = m ? hn * (gamma[bidx * Dd + j] + 1.f) + beta[bidx * Dd + j]
                : hn * (ln_g[j] + 1.f);
    xin[k] = t; q += t * t;
  }
  sh[tid] = q; __syncthreads();
  for (int st = 64; st > 0; st >>= 1) { if (tid < st) sh[tid] += sh[tid + st]; __syncthreads(); }
  float nrm = sqrtf(sh[0]);
  float scl = 22.62741699796952f / fmaxf(nrm, 1e-12f);  // sqrt(512)/||x||
#pragma unroll
  for (int k = 0; k < 4; ++k) {
    int j = tid + 128 * k;
    out[(size_t)tok * ldo + j] = (__bf16)(xin[k] * scl * (norm_g[j] + 1.f));
  }
}

// ---------------------------------------------------------------------------
// Transpose-convert fp32 W[K][N] -> bf16 Wt[Np][Kp] (zero padded)
// ---------------------------------------------------------------------------
__global__ __launch_bounds__(256)
void k_wt(const float* __restrict__ W, __bf16* __restrict__ Wt,
          int K, int N, int Kp, int Np) {
  int idx = blockIdx.x * 256 + threadIdx.x;
  if (idx >= Np * Kp) return;
  int n = idx / Kp, k = idx - n * Kp;
  Wt[idx] = (n < N && k < K) ? (__bf16)W[(size_t)k * N + n] : (__bf16)0.f;
}

// ---------------------------------------------------------------------------
// WMMA bf16 GEMM: C[M][N] = A[M][K] * Wt[N][K]^T   (fp32 accumulate)
// Block 128 threads = 4 wave32 waves; workgroup tile 128x64; K-step 32.
// B tile double-buffered in LDS (async global->LDS DMA when available).
// Each wave computes a 32x64 tile: 8 WMMAs per K-step.
// ---------------------------------------------------------------------------
__global__ __launch_bounds__(128)
void k_gemm(const __bf16* __restrict__ A, const __bf16* __restrict__ Bt,
            float* __restrict__ C, int M, int N, int K, int ldc) {
  __shared__ __attribute__((aligned(16))) __bf16 Bs[2][64][40];  // 80B row stride
  const int tid  = threadIdx.x;
  const int wave = tid >> 5, lane = tid & 31;
  const int half = lane >> 4, l15 = lane & 15;
  const int nb = blockIdx.x * 64, mb = blockIdx.y * 128;
  const int r  = tid >> 1, cb = (tid & 1) * 16;    // B staging coords
  const __bf16* Ap0 = A + (size_t)(mb + wave * 32 + l15) * K;
  const __bf16* Ap1 = Ap0 + (size_t)16 * K;
  const __bf16* Bp  = Bt + (size_t)(nb + r) * K + cb;
  v8f acc[2][4] = {};

  auto stage = [&](int buf, int k0) {
#if ASYNC_LDS
    __builtin_amdgcn_global_load_async_to_lds_b128(
        (AS1 v4i*)(Bp + k0), (AS3 v4i*)&Bs[buf][r][cb], 0, 0);
    __builtin_amdgcn_global_load_async_to_lds_b128(
        (AS1 v4i*)(Bp + k0 + 8), (AS3 v4i*)&Bs[buf][r][cb + 8], 0, 0);
#else
    *(v8bf*)&Bs[buf][r][cb]     = *(const v8bf*)(Bp + k0);
    *(v8bf*)&Bs[buf][r][cb + 8] = *(const v8bf*)(Bp + k0 + 8);
#endif
  };

  stage(0, 0);
  wait_async_lds();
  __syncthreads();

  int cur = 0;
  for (int k0 = 0; k0 < K; k0 += 32) {
    int kn = k0 + 32;
    if (kn < K) {
      stage(cur ^ 1, kn);                      // prefetch next B tile into LDS
      __builtin_prefetch(Ap0 + kn, 0, 1);
      __builtin_prefetch(Ap1 + kn, 0, 1);
    }
    // A fragments: K contiguous per half (lanes<16: K0-15; lanes>=16: K16-31)
    v16bf a0, a1;
    ((v8bf*)&a0)[0] = *(const v8bf*)(Ap0 + k0 + half * 16);
    ((v8bf*)&a0)[1] = *(const v8bf*)(Ap0 + k0 + half * 16 + 8);
    ((v8bf*)&a1)[0] = *(const v8bf*)(Ap1 + k0 + half * 16);
    ((v8bf*)&a1)[1] = *(const v8bf*)(Ap1 + k0 + half * 16 + 8);
#pragma unroll
    for (int t = 0; t < 4; ++t) {
      // B fragment: lane holds column n = 16t+l15; K contiguous per half
      v16bf bfr;
      const __bf16* bsp = &Bs[cur][16 * t + l15][half * 16];
      ((v8bf*)&bfr)[0] = *(const v8bf*)bsp;
      ((v8bf*)&bfr)[1] = *(const v8bf*)(bsp + 8);
      acc[0][t] = __builtin_amdgcn_wmma_f32_16x16x32_bf16(false, a0, false, bfr,
                                                          (short)0, acc[0][t], false, false);
      acc[1][t] = __builtin_amdgcn_wmma_f32_16x16x32_bf16(false, a1, false, bfr,
                                                          (short)0, acc[1][t], false, false);
    }
    wait_async_lds();
    __syncthreads();
    cur ^= 1;
  }
#pragma unroll
  for (int pr = 0; pr < 2; ++pr) {
#pragma unroll
    for (int t = 0; t < 4; ++t) {
      int col = nb + 16 * t + l15;
      if (col < N) {
#pragma unroll
        for (int v = 0; v < 8; ++v) {
          int row = mb + wave * 32 + pr * 16 + v + half * 8;  // C layout
          C[(size_t)row * ldc + col] = acc[pr][t][v];
        }
      }
    }
  }
}

// ---------------------------------------------------------------------------
// Unpack qkv GEMM output -> Q (scaled), K row-major [bh][s][dh], V transposed
// ---------------------------------------------------------------------------
__global__ __launch_bounds__(256)
void k_pack(const float* __restrict__ C, __bf16* __restrict__ Qb,
            __bf16* __restrict__ Kb, __bf16* __restrict__ Vt) {
  int idx = blockIdx.x * 256 + threadIdx.x;
  if (idx >= Bb * Hh * Ss * DHd) return;
  int d = idx & 63;
  int s = (idx >> 6) & (Ss - 1);
  int h = (idx >> 17) & 7;
  int b = idx >> 20;
  size_t tok = (size_t)b * Ss + s;
  const float* base = C + tok * (3 * Hh * DHd) + h * DHd + d;
  float q = base[0] * 0.125f;                      // DIM_HEAD^-0.5
  float k = base[Hh * DHd];
  float v = base[2 * Hh * DHd];
  size_t bh = (size_t)(b * Hh + h);
  Qb[(bh * Ss + s) * DHd + d] = (__bf16)q;
  Kb[(bh * Ss + s) * DHd + d] = (__bf16)k;
  Vt[(bh * DHd + d) * Ss + s] = (__bf16)v;
}

// ---------------------------------------------------------------------------
// Flash attention with softcap + causal mask. Grid: (S/64, B*H), 4 waves.
// ---------------------------------------------------------------------------
__global__ __launch_bounds__(128)
void k_attn(const __bf16* __restrict__ Q, const __bf16* __restrict__ Km,
            const __bf16* __restrict__ Vt, __bf16* __restrict__ O) {
  __shared__ __attribute__((aligned(16))) __bf16 Ps[4][16][72];  // per-wave P tile
  const int tid = threadIdx.x;
  const int w = tid >> 5, lane = tid & 31;
  const int half = lane >> 4, l15 = lane & 15;
  const int bh = blockIdx.y;
  const int b = bh >> 3, h = bh & 7;
  const int q0b = blockIdx.x * 64;
  const int q0 = q0b + w * 16;
  const __bf16* Qp = Q + ((size_t)bh * Ss + q0 + l15) * DHd;
  v16bf qa[2];
#pragma unroll
  for (int ks = 0; ks < 2; ++ks) {
    ((v8bf*)&qa[ks])[0] = *(const v8bf*)(Qp + 32 * ks + half * 16);
    ((v8bf*)&qa[ks])[1] = *(const v8bf*)(Qp + 32 * ks + half * 16 + 8);
  }
  v8f acc[4] = {};
  float mrow[8], lrow[8];
#pragma unroll
  for (int v = 0; v < 8; ++v) { mrow[v] = -3.4e38f; lrow[v] = 0.f; }

  for (int j0 = 0; j0 <= q0b; j0 += 64) {          // causal key-tile loop
    v8f s[4] = {};
#pragma unroll
    for (int t = 0; t < 4; ++t) {
      const __bf16* Kp = Km + ((size_t)bh * Ss + j0 + 16 * t + l15) * DHd;
#pragma unroll
      for (int ks = 0; ks < 2; ++ks) {
        v16bf kf;                                   // B operand = K^T column
        ((v8bf*)&kf)[0] = *(const v8bf*)(Kp + 32 * ks + half * 16);
        ((v8bf*)&kf)[1] = *(const v8bf*)(Kp + 32 * ks + half * 16 + 8);
        s[t] = __builtin_amdgcn_wmma_f32_16x16x32_bf16(false, qa[ks], false, kf,
                                                       (short)0, s[t], false, false);
      }
    }
    // softcap + causal mask + row max
    float rm[8];
#pragma unroll
    for (int v = 0; v < 8; ++v) rm[v] = -3.4e38f;
#pragma unroll
    for (int t = 0; t < 4; ++t) {
      int key = j0 + 16 * t + l15;
#pragma unroll
      for (int v = 0; v < 8; ++v) {
        int qr = q0 + v + half * 8;
        float x = tanhf(s[t][v] * 0.02f) * 50.f;
        if (key > qr) x = -3.4e38f;
        s[t][v] = x;
        rm[v] = fmaxf(rm[v], x);
      }
    }
#pragma unroll
    for (int v = 0; v < 8; ++v)
#pragma unroll
      for (int off = 1; off < 16; off <<= 1)
        rm[v] = fmaxf(rm[v], __shfl_xor(rm[v], off, 32));
    float sc[8], rs[8];
#pragma unroll
    for (int v = 0; v < 8; ++v) {
      float mn = fmaxf(mrow[v], rm[v]);
      sc[v] = __expf(mrow[v] - mn);
      mrow[v] = mn;
      rs[v] = 0.f;
    }
#pragma unroll
    for (int t = 0; t < 4; ++t)
#pragma unroll
      for (int v = 0; v < 8; ++v) {
        float p = __expf(s[t][v] - mrow[v]);
        s[t][v] = p;
        rs[v] += p;
      }
#pragma unroll
    for (int v = 0; v < 8; ++v) {
#pragma unroll
      for (int off = 1; off < 16; off <<= 1)
        rs[v] += __shfl_xor(rs[v], off, 32);
      lrow[v] = lrow[v] * sc[v] + rs[v];
    }
    // rescale accumulators; stash P (C layout) to LDS
#pragma unroll
    for (int t = 0; t < 4; ++t)
#pragma unroll
      for (int v = 0; v < 8; ++v) {
        acc[t][v] *= sc[v];
        Ps[w][v + half * 8][16 * t + l15] = (__bf16)s[t][v];
      }
    __syncthreads();
#pragma unroll
    for (int ks = 0; ks < 2; ++ks) {
      v16bf pa;                                     // A operand from LDS
      ((v8bf*)&pa)[0] = *(const v8bf*)&Ps[w][l15][32 * ks + half * 16];
      ((v8bf*)&pa)[1] = *(const v8bf*)&Ps[w][l15][32 * ks + half * 16 + 8];
#pragma unroll
      for (int t = 0; t < 4; ++t) {
        const __bf16* Vp = Vt + ((size_t)bh * DHd + 16 * t + l15) * Ss
                              + j0 + 32 * ks + half * 16;
        v16bf vf;
        ((v8bf*)&vf)[0] = *(const v8bf*)Vp;
        ((v8bf*)&vf)[1] = *(const v8bf*)(Vp + 8);
        acc[t] = __builtin_amdgcn_wmma_f32_16x16x32_bf16(false, pa, false, vf,
                                                         (short)0, acc[t], false, false);
      }
    }
    __syncthreads();
  }
  // normalize and scatter to token-major [4096][512] bf16 (out-proj A matrix)
#pragma unroll
  for (int t = 0; t < 4; ++t)
#pragma unroll
    for (int v = 0; v < 8; ++v) {
      int qr = q0 + v + half * 8;
      float o = acc[t][v] / lrow[v];
      O[((size_t)(b * Ss + qr)) * Dd + h * DHd + 16 * t + l15] = (__bf16)o;
    }
}

// ---------------------------------------------------------------------------
// GEGLU activation: act = gelu_exact(g) * a, zero-padded to FFKP, bf16 out.
// ---------------------------------------------------------------------------
__global__ __launch_bounds__(256)
void k_ffact(const float* __restrict__ C, const float* __restrict__ b1,
             __bf16* __restrict__ out) {
  int idx = blockIdx.x * 256 + threadIdx.x;
  if (idx >= Mm * FFKP) return;
  int tok = idx / FFKP, j = idx - tok * FFKP;
  float val = 0.f;
  if (j < FF) {
    float a = C[(size_t)tok * FF2P + j] + b1[j];
    float g = C[(size_t)tok * FF2P + FF + j] + b1[FF + j];
    float ge = 0.5f * g * (1.f + erff(g * 0.70710678118654752f));
    val = ge * a;
  }
  out[idx] = (__bf16)val;
}

// ---------------------------------------------------------------------------
// Gated residual: x += where(mod, (C+bias)*gate, (C+bias)*lscale)
// ---------------------------------------------------------------------------
__global__ __launch_bounds__(256)
void k_gated(const float* __restrict__ C, const float* __restrict__ bias,
             const unsigned char* __restrict__ mod, const float* __restrict__ gate,
             const float* __restrict__ lscale, float* __restrict__ x, int ldc) {
  int idx = blockIdx.x * 256 + threadIdx.x;
  if (idx >= Mm * Dd) return;
  int tok = idx >> 9, j = idx & (Dd - 1);
  int b = tok >> 11;
  float v = C[(size_t)tok * ldc + j];
  if (bias) v += bias[j];
  v = mod[tok] ? v * gate[b * Dd + j] : v * lscale[j];
  x[idx] += v;
}

// ---------------------------------------------------------------------------
// Final rmsnorm -> fp32 output
// ---------------------------------------------------------------------------
__global__ __launch_bounds__(128)
void k_final(const float* __restrict__ x, const float* __restrict__ g,
             float* __restrict__ out) {
  __shared__ float sh[128];
  const int tok = blockIdx.x, tid = threadIdx.x;
  const float* xp = x + (size_t)tok * Dd;
  float v[4]; float s2 = 0.f;
#pragma unroll
  for (int k = 0; k < 4; ++k) { v[k] = xp[tid + 128 * k]; s2 += v[k] * v[k]; }
  sh[tid] = s2; __syncthreads();
  for (int st = 64; st > 0; st >>= 1) { if (tid < st) sh[tid] += sh[tid + st]; __syncthreads(); }
  float nrm = sqrtf(sh[0]);
  float scl = 22.62741699796952f / fmaxf(nrm, 1e-12f);
#pragma unroll
  for (int k = 0; k < 4; ++k) {
    int j = tid + 128 * k;
    out[(size_t)tok * Dd + j] = v[k] * scl * (g[j] + 1.f);
  }
}

// ---------------------------------------------------------------------------
extern "C" void kernel_launch(void* const* d_in, const int* in_sizes, int n_in,
                              void* d_out, int out_size, void* d_ws, size_t ws_size,
                              hipStream_t stream) {
  (void)in_sizes; (void)n_in; (void)out_size; (void)ws_size;
  const float* x_in        = (const float*)d_in[0];
  const float* times       = (const float*)d_in[1];
  /* d_in[2] = attn_mask (causal, implemented directly) */
  const unsigned char* mod = (const unsigned char*)d_in[3];
  const float* fourier_w   = (const float*)d_in[4];
  const float* time_w      = (const float*)d_in[5];
  const float* time_b      = (const float*)d_in[6];
  const float* aw_ln_g     = (const float*)d_in[7];
  const float* aw_scale    = (const float*)d_in[8];
  const float* aw_film_w   = (const float*)d_in[9];
  const float* aw_film_b   = (const float*)d_in[10];
  const float* aw_zero_w   = (const float*)d_in[11];
  const float* aw_zero_b   = (const float*)d_in[12];
  const float* attn_norm_g = (const float*)d_in[13];
  const float* qkv_w       = (const float*)d_in[14];
  const float* out_w       = (const float*)d_in[15];
  const float* fw_ln_g     = (const float*)d_in[16];
  const float* fw_scale    = (const float*)d_in[17];
  const float* fw_film_w   = (const float*)d_in[18];
  const float* fw_film_b   = (const float*)d_in[19];
  const float* fw_zero_w   = (const float*)d_in[20];
  const float* fw_zero_b   = (const float*)d_in[21];
  const float* ff_norm_g   = (const float*)d_in[22];
  const float* ff_w1       = (const float*)d_in[23];
  const float* ff_b1       = (const float*)d_in[24];
  const float* ff_w2       = (const float*)d_in[25];
  const float* ff_b2       = (const float*)d_in[26];
  const float* final_g     = (const float*)d_in[27];

  // Workspace carve-up (~80 MB)
  char* p = (char*)d_ws;
  auto take = [&](size_t bytes) {
    char* r = p; p += (bytes + 255) & ~(size_t)255; return r;
  };
  float*  xf    = (float*)take((size_t)Mm * Dd * 4);
  float*  cond  = (float*)take((size_t)Bb * DCc * 4);
  float*  gamma = (float*)take((size_t)Bb * Dd * 4);
  float*  beta  = (float*)take((size_t)Bb * Dd * 4);
  float*  gate  = (float*)take((size_t)Bb * Dd * 4);
  __bf16* Abf   = (__bf16*)take((size_t)Mm * FFKP * 2);
  float*  C32   = (float*)take((size_t)Mm * FF2P * 4);
  __bf16* Wt    = (__bf16*)take((size_t)FF2P * Dd * 2);
  __bf16* Qb    = (__bf16*)take((size_t)Bb * Hh * Ss * DHd * 2);
  __bf16* Kb    = (__bf16*)take((size_t)Bb * Hh * Ss * DHd * 2);
  __bf16* Vt    = (__bf16*)take((size_t)Bb * Hh * Ss * DHd * 2);

  (void)hipMemcpyAsync(xf, x_in, (size_t)Mm * Dd * 4, hipMemcpyDeviceToDevice, stream);
  k_cond<<<(Bb * DCc) / 128, 128, 0, stream>>>(times, fourier_w, time_w, time_b, cond);

  for (int l = 0; l < 4; ++l) {
    // ---------------- attention branch ----------------
    k_film<<<(Bb * 1536) / 128, 128, 0, stream>>>(
        cond, aw_film_w + (size_t)l * DCc * 1024, aw_film_b + (size_t)l * 1024,
        aw_zero_w + (size_t)l * DCc * Dd, aw_zero_b + (size_t)l * Dd,
        gamma, beta, gate);
    k_adaln<<<Mm, 128, 0, stream>>>(xf, mod, gamma, beta,
                                    aw_ln_g + (size_t)l * Dd,
                                    attn_norm_g + (size_t)l * Dd, Abf, Dd);
    {   // qkv projection: [4096,512] x [512,1536]
      int K = Dd, N = 3 * Hh * DHd;
      k_wt<<<((size_t)N * K + 255) / 256, 256, 0, stream>>>(
          qkv_w + (size_t)l * K * N, Wt, K, N, K, N);
      k_gemm<<<dim3(N / 64, Mm / 128), 128, 0, stream>>>(Abf, Wt, C32, Mm, N, K, N);
    }
    k_pack<<<((size_t)Bb * Hh * Ss * DHd + 255) / 256, 256, 0, stream>>>(C32, Qb, Kb, Vt);
    k_attn<<<dim3(Ss / 64, Bb * Hh), 128, 0, stream>>>(Qb, Kb, Vt, Abf);
    {   // out projection: [4096,512] x [512,512]
      int K = Dd, N = Dd;
      k_wt<<<((size_t)N * K + 255) / 256, 256, 0, stream>>>(
          out_w + (size_t)l * K * N, Wt, K, N, K, N);
      k_gemm<<<dim3(N / 64, Mm / 128), 128, 0, stream>>>(Abf, Wt, C32, Mm, N, K, N);
    }
    k_gated<<<(Mm * Dd + 255) / 256, 256, 0, stream>>>(
        C32, (const float*)nullptr, mod, gate, aw_scale + (size_t)l * Dd, xf, Dd);

    // ---------------- feed-forward branch ----------------
    k_film<<<(Bb * 1536) / 128, 128, 0, stream>>>(
        cond, fw_film_w + (size_t)l * DCc * 1024, fw_film_b + (size_t)l * 1024,
        fw_zero_w + (size_t)l * DCc * Dd, fw_zero_b + (size_t)l * Dd,
        gamma, beta, gate);
    k_adaln<<<Mm, 128, 0, stream>>>(xf, mod, gamma, beta,
                                    fw_ln_g + (size_t)l * Dd,
                                    ff_norm_g + (size_t)l * Dd, Abf, Dd);
    {   // ff1: [4096,512] x [512,2730] (N padded to 2752)
      int K = Dd, N = FF2, Np = FF2P;
      k_wt<<<((size_t)Np * K + 255) / 256, 256, 0, stream>>>(
          ff_w1 + (size_t)l * K * N, Wt, K, N, K, Np);
      k_gemm<<<dim3(Np / 64, Mm / 128), 128, 0, stream>>>(Abf, Wt, C32, Mm, N, K, Np);
    }
    k_ffact<<<((size_t)Mm * FFKP + 255) / 256, 256, 0, stream>>>(
        C32, ff_b1 + (size_t)l * FF2, Abf);
    {   // ff2: [4096,1365(pad 1376)] x [1365,512]
      int K = FF, Kp = FFKP, N = Dd;
      k_wt<<<((size_t)N * Kp + 255) / 256, 256, 0, stream>>>(
          ff_w2 + (size_t)l * K * N, Wt, K, N, Kp, N);
      k_gemm<<<dim3(N / 64, Mm / 128), 128, 0, stream>>>(Abf, Wt, C32, Mm, N, Kp, N);
    }
    k_gated<<<(Mm * Dd + 255) / 256, 256, 0, stream>>>(
        C32, ff_b2 + (size_t)l * Dd, mod, gate, fw_scale + (size_t)l * Dd, xf, Dd);
  }
  k_final<<<Mm, 128, 0, stream>>>(xf, final_g, (float*)d_out);
}
